// MyFunctionBlockSeriell_43276090474801
// MI455X (gfx1250) — matrix operand order repaired and checked by reading, not compiled, not run on hardware
//
#include <hip/hip_runtime.h>
#include <hip/hip_bf16.h>
#include <math.h>

// ---- CDNA5 WMMA vector types ----
typedef __attribute__((ext_vector_type(16))) __bf16 v16bf;
typedef __attribute__((ext_vector_type(8)))  __bf16 v8bf;
typedef __attribute__((ext_vector_type(4)))  __bf16 v4bf;
typedef __attribute__((ext_vector_type(8)))  float  v8f;

#define FDIM 128
#define ROWS_PER_TILE 16
#define WAVES_PER_BLOCK 8
#define PITCH_BYTES 528                              // 256 bf16 (512B) + 16B pad -> conflict-free ds_load_b128
#define TILE_BYTES (ROWS_PER_TILE * PITCH_BYTES)     // 8448
#define WAVE_LDS   (TILE_BYTES + 2 * 16 * 4)         // + dsum[16] + dprod[16] = 8576 (16B aligned)

__global__ __launch_bounds__(256) void fused_symreg_kernel(
    const float* __restrict__ x,
    const float* __restrict__ W_prod, const float* __restrict__ b_prod,
    const float* __restrict__ W_sum,  const float* __restrict__ b_sum,
    const float* __restrict__ w_dln,  const float* __restrict__ W_ln,
    const float* __restrict__ w_dsin, const float* __restrict__ W_sin,
    const float* __restrict__ w_dcos, const float* __restrict__ W_cos,
    const float* __restrict__ w_de,   const float* __restrict__ W_e,
    const float* __restrict__ w_dtanh,const float* __restrict__ W_tanh,
    float* __restrict__ out, int num_rows)
{
    __shared__ __align__(16) unsigned char smem[WAVES_PER_BLOCK * WAVE_LDS];

    const int tid  = (int)threadIdx.x;
    const int wave = tid >> 5;
    const int lane = tid & 31;
    const int n    = lane & 15;   // N index for B / C / D, also A row (M) index
    const int sub  = lane >> 4;   // half-select within K chunk

    unsigned char* wbase = smem + wave * WAVE_LDS;
    float* dsum  = (float*)(wbase + TILE_BYTES);
    float* dprod = dsum + 16;

    // ---- uniform scalar parameters ----
    const float c_bprod = b_prod[0];
    const float c_wxp   = W_sum[FDIM];      // weight applied to x_prod in SumBlock
    const float c_bsum  = b_sum[0];
    const float c_wdln  = w_dln[0],   c_ln0 = W_ln[0],   c_ln1 = W_ln[1];
    const float c_wdsin = w_dsin[0],  c_sn0 = W_sin[0],  c_sn1 = W_sin[1];
    const float c_wdcos = w_dcos[0],  c_cs0 = W_cos[0],  c_cs1 = W_cos[1];
    const float c_wde   = w_de[0],    c_e0  = W_e[0],    c_e1  = W_e[1];
    const float c_wdth  = w_dtanh[0], c_th0 = W_tanh[0], c_th1 = W_tanh[1];

    // ---- loop-invariant B fragments -------------------------------------
    // A rows are [x(K=0..127) | log|x|(K=128..255)] -> 8 chunks of K=32.
    // B column 0 carries W_sum[0:128] on the x half; column 1 carries W_prod
    // on the log half; all other columns zero. 16-bit B 32x16 layout:
    // lanes 0-15 hold K={0..7,16..23} (N = lane), lanes 16-31 hold K={8..15,24..31}.
    v16bf bfrag[8];
    #pragma unroll
    for (int c = 0; c < 8; ++c) {
        #pragma unroll
        for (int j = 0; j < 16; ++j) {
            const int kk = 32 * c + 8 * sub + j + (j >= 8 ? 8 : 0);
            float w = 0.0f;
            if (kk < FDIM) { if (n == 0) w = W_sum[kk]; }
            else           { if (n == 1) w = W_prod[kk - FDIM]; }
            bfrag[c][j] = (__bf16)w;
        }
    }

    const int waves_total = (int)gridDim.x * WAVES_PER_BLOCK;
    const int wave_gid    = (int)blockIdx.x * WAVES_PER_BLOCK + wave;
    const int num_tiles   = num_rows / ROWS_PER_TILE;

    for (int tile = wave_gid; tile < num_tiles; tile += waves_total) {
        const int row0 = tile * ROWS_PER_TILE;

        // ---- stage tile into LDS as bf16 [x | log|x|], coalesced loads ----
        const float4* src = (const float4*)(x + (size_t)row0 * FDIM);
        #pragma unroll
        for (int r = 0; r < ROWS_PER_TILE; ++r) {
            float4 v = src[r * (FDIM / 4) + lane];
            v4bf xe, le;
            xe[0] = (__bf16)v.x; xe[1] = (__bf16)v.y;
            xe[2] = (__bf16)v.z; xe[3] = (__bf16)v.w;
            le[0] = (__bf16)__logf(fabsf(v.x));
            le[1] = (__bf16)__logf(fabsf(v.y));
            le[2] = (__bf16)__logf(fabsf(v.z));
            le[3] = (__bf16)__logf(fabsf(v.w));
            __bf16* rowp = (__bf16*)(wbase + r * PITCH_BYTES);
            *(v4bf*)(rowp + 4 * lane)        = xe;
            *(v4bf*)(rowp + FDIM + 4 * lane) = le;
        }
        // prefetch ALL 64 x 128B cachelines of the next tile into near caches
        if (tile + waves_total < num_tiles) {
            const char* nxt = (const char*)(x + (size_t)(tile + waves_total) * ROWS_PER_TILE * FDIM);
            __builtin_prefetch(nxt + lane * 128,        0, 3);
            __builtin_prefetch(nxt + 4096 + lane * 128, 0, 3);
        }
        __syncthreads();

        // ---- D = A x B over K=256 : 8 x v_wmma_f32_16x16x32_bf16 ----
        // Software-pipelined A-fragment loads: issue loads for chunk c+1
        // before the WMMA of chunk c so LDS latency overlaps matrix issue.
        // A 16-bit 16x32 layout: lane<16 (row n) holds K={0..7,16..23},
        // lane>=16 holds K={8..15,24..31} of each chunk.
        const unsigned char* arow = wbase + n * PITCH_BYTES + 16 * sub;
        v8f acc = {};
        v8bf lo = *(const v8bf*)(arow);
        v8bf hi = *(const v8bf*)(arow + 32);
        #pragma unroll
        for (int c = 0; c < 8; ++c) {
            v8bf lo_n, hi_n;
            if (c < 7) {
                lo_n = *(const v8bf*)(arow + (c + 1) * 64);
                hi_n = *(const v8bf*)(arow + (c + 1) * 64 + 32);
            }
            v16bf a;
            #pragma unroll
            for (int i = 0; i < 8; ++i) { a[i] = lo[i]; a[8 + i] = hi[i]; }
            acc = __builtin_amdgcn_wmma_f32_16x16x32_bf16(
                      /*neg_a=*/false, a, /*neg_b=*/false, bfrag[c],
                      /*c_mod=*/(short)0, acc, /*reuse_a=*/false, /*reuse_b=*/false);
            lo = lo_n; hi = hi_n;
        }

        // ---- scatter the two useful D columns (N=0: x-dot, N=1: log-dot) ----
        // C/D layout: VGPR r -> M = r (lanes 0-15) or M = 8+r (lanes 16-31).
        if (n == 0) {
            #pragma unroll
            for (int r = 0; r < 8; ++r) dsum[8 * sub + r] = acc[r];
        } else if (n == 1) {
            #pragma unroll
            for (int r = 0; r < 8; ++r) dprod[8 * sub + r] = acc[r];
        }
        __syncthreads();

        // ---- per-row serial tail on lanes 0..15 ----
        if (lane < ROWS_PER_TILE) {
            const float xp = __expf(dprod[lane] + c_bprod);         // ProductBlock
            float h = dsum[lane] + xp * c_wxp + c_bsum;             // SumBlock
            float d;
            d = fmaxf(__logf(fabsf(h)) * c_wdln, 0.0f); h = d * c_ln0 + h * c_ln1;
            d = fmaxf(__sinf(h) * c_wdsin,       0.0f); h = d * c_sn0 + h * c_sn1;
            d = fmaxf(__cosf(h) * c_wdcos,       0.0f); h = d * c_cs0 + h * c_cs1;
            d = fmaxf(__expf(h) * c_wde,         0.0f); h = d * c_e0  + h * c_e1;
            d = fmaxf(tanhf(h) * c_wdth,         0.0f); h = d * c_th0 + h * c_th1;
            out[row0 + lane] = h;
        }
        __syncthreads();   // protect LDS tile before next iteration overwrites it
    }
}

extern "C" void kernel_launch(void* const* d_in, const int* in_sizes, int n_in,
                              void* d_out, int out_size, void* d_ws, size_t ws_size,
                              hipStream_t stream) {
    const float* x       = (const float*)d_in[0];
    const float* W_prod  = (const float*)d_in[1];
    const float* b_prod  = (const float*)d_in[2];
    const float* W_sum   = (const float*)d_in[3];
    const float* b_sum   = (const float*)d_in[4];
    const float* w_dln   = (const float*)d_in[5];
    const float* W_ln    = (const float*)d_in[6];
    const float* w_dsin  = (const float*)d_in[7];
    const float* W_sin   = (const float*)d_in[8];
    const float* w_dcos  = (const float*)d_in[9];
    const float* W_cos   = (const float*)d_in[10];
    const float* w_de    = (const float*)d_in[11];
    const float* W_e     = (const float*)d_in[12];
    const float* w_dtanh = (const float*)d_in[13];
    const float* W_tanh  = (const float*)d_in[14];

    const int num_rows = in_sizes[0] / FDIM;   // B = 1048576

    dim3 grid(2048), block(256);
    fused_symreg_kernel<<<grid, block, 0, stream>>>(
        x, W_prod, b_prod, W_sum, b_sum,
        w_dln, W_ln, w_dsin, W_sin, w_dcos, W_cos,
        w_de, W_e, w_dtanh, W_tanh,
        (float*)d_out, num_rows);
}
